// TCPGA_80238579024261
// MI455X (gfx1250) — matrix operand order repaired and checked by reading, not compiled, not run on hardware
//
#include <hip/hip_runtime.h>

typedef __attribute__((ext_vector_type(16))) __bf16 v16bf;
typedef __attribute__((ext_vector_type(8)))  float  v8f;
typedef __attribute__((ext_vector_type(4)))  unsigned int uv4;
typedef __attribute__((ext_vector_type(8)))  int          iv8;
typedef __attribute__((ext_vector_type(4)))  int          iv4;
typedef unsigned int   u32;
typedef unsigned short u16;
typedef unsigned long long u64;

#define DIMM 512
#define DD   256
#define NN   4096
#define BB   16
#define KQ   32

union Frag { v16bf bf; u32 u[8]; };
union Acc  { v8f v; float f[8]; };

__device__ __forceinline__ u16 f2bf(float x){
  u32 u = __float_as_uint(x);
  u += 0x7FFFu + ((u >> 16) & 1u);     // round-to-nearest-even
  return (u16)(u >> 16);
}

// --- Tensor Data Mover: async 2D tile (rows x cols of 16-bit elems) global -> LDS.
// D# per CDNA5 ISA 8.3/8.4: group0 = {count, lds_addr, global_addr[56:0], type=2},
// group1 = {data_size=1(2B), tensor_dim0/1, tile_dim0/1, tensor_dim0_stride}.
// Trailing groups zero (2D tensor). Issued once per wave (EXEC ignored); TENSORcnt tracks.
__device__ __forceinline__ void tdm_load_2d_b16(const u16* gptr, u32 lds_off,
                                                u32 tensor_d0, u32 tensor_d1,
                                                u32 stride_d0, u32 tile_d0, u32 tile_d1){
  u64 ga = (u64)(const void*)gptr;
  uv4 g0;
  g0.x = 1u;                                    // count = 1 (valid descriptor)
  g0.y = lds_off;                               // LDS byte address of tile start
  g0.z = (u32)ga;                               // global_addr[31:0]
  g0.w = (u32)((ga >> 32) & 0x1FFFFFFull) | (2u << 30);  // global_addr[56:32] | type=2
  iv8 g1;
  g1[0] = (int)(1u << 16);                      // workgroup_mask=0, data_size=1 (2 bytes)
  g1[1] = (int)((tensor_d0 & 0xFFFFu) << 16);   // atomic_barrier_addr=0 | tensor_dim0[15:0]
  g1[2] = (int)(((tensor_d0 >> 16) & 0xFFFFu) | ((tensor_d1 & 0xFFFFu) << 16));
  g1[3] = (int)(((tensor_d1 >> 16) & 0xFFFFu) | ((tile_d0 & 0xFFFFu) << 16));
  g1[4] = (int)(tile_d1 & 0xFFFFu);             // tile_dim1 | tile_dim2=0
  g1[5] = (int)stride_d0;                       // tensor_dim0_stride[31:0]
  g1[6] = 0;                                    // stride[47:32] | dim1_stride[15:0]
  g1[7] = 0;
  iv4 z4 = {0, 0, 0, 0};
  iv8 z8 = {0, 0, 0, 0, 0, 0, 0, 0};
  __builtin_amdgcn_tensor_load_to_lds(g0, g1, z4, z4, z8, 0);
}

// A-fragment: 16x32 bf16, A staged in LDS row-major [m][32 k] (16 dwords/row)
__device__ __forceinline__ void load_afrag(Frag &a, const u32* As, int mbase, int lane){
  int m  = mbase + (lane & 15);
  int hi = (lane >> 4) * 8;
  #pragma unroll
  for (int v = 0; v < 8; ++v){
    int kd = v + hi;   // dword holding the K pair
    a.u[v] = As[m * 16 + kd];
  }
}

// B-fragment: 32x16 bf16, B staged in LDS as [n][32 k] (16 dwords/row)
__device__ __forceinline__ void load_bfrag(Frag &b, const u32* Bs, int nbase, int lane){
  int n  = nbase + (lane & 15);
  int hi = (lane >> 4) * 8;
  #pragma unroll
  for (int v = 0; v < 8; ++v)
    b.u[v] = Bs[n * 16 + v + hi];
}

__device__ __forceinline__ v8f wmma_bf16(const Frag &a, const Frag &b, v8f c){
  return __builtin_amdgcn_wmma_f32_16x16x32_bf16(false, a.bf, false, b.bf,
                                                 (short)0, c, false, false);
}

// fp32 -> bf16 convert (weights)
__global__ __launch_bounds__(256)
void cvt_kernel(const float* __restrict__ in, u16* __restrict__ out, int n){
  int i = blockIdx.x * 256 + threadIdx.x;
  if (i < n) out[i] = f2bf(in[i]);
}

// Fused K/V projection: Kp = E@Wk^T, V = E@Wv^T. One pass over E (the HBM-dominant
// input). Weight tiles (256x32 bf16) are DMA'd by the TDM while all waves convert
// the E tile fp32->bf16; wave 0 closes the DMA with s_wait_tensorcnt before the
// workgroup barrier.
__global__ __launch_bounds__(256)
void proj_kv_kernel(const float* __restrict__ E, const u16* __restrict__ Wk,
                    const u16* __restrict__ Wv, u16* __restrict__ Kp,
                    u16* __restrict__ Vout){
  __shared__ u32 As[64 * 16];     // 64 rows x 32 bf16
  __shared__ u32 Bk[256 * 16];    // Wk tile: 256 cols x 32 bf16
  __shared__ u32 Bv[256 * 16];    // Wv tile
  const int tid = threadIdx.x, lane = tid & 31, wid = tid >> 5;
  const int mw = wid & 3, nw = wid >> 2;
  const int rowBase = blockIdx.x * 64;
  u16* As16 = (u16*)As;
  const u32 bkOff = (u32)(size_t)(void*)Bk;
  const u32 bvOff = (u32)(size_t)(void*)Bv;

  Acc accK[8], accV[8];
  #pragma unroll
  for (int t = 0; t < 8; ++t)
    #pragma unroll
    for (int j = 0; j < 8; ++j){ accK[t].f[j] = 0.f; accV[t].f[j] = 0.f; }

  for (int kk = 0; kk < DIMM; kk += 32){
    if (wid == 0){   // one TDM issue per workgroup for each weight tile
      tdm_load_2d_b16(Wk + kk, bkOff, DIMM, DD, DIMM, 32, DD);
      tdm_load_2d_b16(Wv + kk, bvOff, DIMM, DD, DIMM, 32, DD);
    }
    for (int i = tid; i < 64 * 32; i += 256){
      int r = i >> 5, k = i & 31;
      As16[i] = f2bf(E[(rowBase + r) * DIMM + kk + k]);
    }
    if (kk + 32 < DIMM)   // prefetch next E k-slice (global_prefetch_b8)
      __builtin_prefetch(&E[(rowBase + (tid & 63)) * DIMM + kk + 32], 0, 0);
    if (wid == 0) __builtin_amdgcn_s_wait_tensorcnt(0);
    __syncthreads();
    Frag a; load_afrag(a, As, mw * 16, lane);
    #pragma unroll
    for (int t = 0; t < 8; ++t){
      Frag bfr; load_bfrag(bfr, Bk, nw * 128 + t * 16, lane);
      accK[t].v = wmma_bf16(a, bfr, accK[t].v);
    }
    #pragma unroll
    for (int t = 0; t < 8; ++t){
      Frag bfr; load_bfrag(bfr, Bv, nw * 128 + t * 16, lane);
      accV[t].v = wmma_bf16(a, bfr, accV[t].v);
    }
    __syncthreads();
  }

  const int hi = lane >> 4;
  #pragma unroll
  for (int t = 0; t < 8; ++t){
    int col = nw * 128 + t * 16 + (lane & 15);
    #pragma unroll
    for (int v = 0; v < 8; ++v){
      int row = rowBase + mw * 16 + v + hi * 8;
      Kp[(size_t)row * DD + col]   = f2bf(accK[t].f[v]);
      Vout[(size_t)row * DD + col] = f2bf(accV[t].f[v]);
    }
  }
}

// out[row, col] = sum_k X[row,k] * W[col,k]; compile-time in/out precision.
template<int XF32, int OUTF32>
__global__ __launch_bounds__(256)
void gemm_xwT(const void* __restrict__ Xv, const u16* __restrict__ W,
              void* __restrict__ outv, int innerK, int nout){
  __shared__ u32 As[64 * 16];
  __shared__ u32 Bs[256 * 16];
  const int tid = threadIdx.x, lane = tid & 31, wid = tid >> 5;
  const int mw = wid & 3, nw = wid >> 2;
  const int rowBase = blockIdx.x * 64;
  const int nBase   = blockIdx.y * 256;
  const float* Xf = (const float*)Xv;
  const u16*   Xb = (const u16*)Xv;
  const u32*   Wu = (const u32*)W;
  u16* As16 = (u16*)As;
  const int kdw = innerK >> 1;

  Acc acc[8];
  #pragma unroll
  for (int t = 0; t < 8; ++t)
    #pragma unroll
    for (int j = 0; j < 8; ++j) acc[t].f[j] = 0.f;

  for (int kk = 0; kk < innerK; kk += 32){
    for (int i = tid; i < 64 * 32; i += 256){
      int r = i >> 5, k = i & 31;
      int gi = (rowBase + r) * innerK + kk + k;
      As16[i] = XF32 ? f2bf(Xf[gi]) : Xb[gi];
    }
    for (int i = tid; i < 256 * 16; i += 256){
      int r = i >> 4, kd = i & 15;
      Bs[i] = Wu[(nBase + r) * kdw + (kk >> 1) + kd];
    }
    __syncthreads();
    Frag a; load_afrag(a, As, mw * 16, lane);
    #pragma unroll
    for (int t = 0; t < 8; ++t){
      Frag bfr; load_bfrag(bfr, Bs, nw * 128 + t * 16, lane);
      acc[t].v = wmma_bf16(a, bfr, acc[t].v);
    }
    __syncthreads();
  }

  float* Of = (float*)outv;
  u16*   Ob = (u16*)outv;
  const int hi = lane >> 4;
  #pragma unroll
  for (int t = 0; t < 8; ++t){
    int col = nBase + nw * 128 + t * 16 + (lane & 15);
    #pragma unroll
    for (int v = 0; v < 8; ++v){
      int row = rowBase + mw * 16 + v + hi * 8;
      if (OUTF32) Of[(size_t)row * nout + col] = acc[t].f[v];
      else        Ob[(size_t)row * nout + col] = f2bf(acc[t].f[v]);
    }
  }
}

// scores[b, m, n] = (Q[b,m,:] . Kp[b,n,:]) * SCALE + log(clip(P[b,n])) -> A region.
// Kp tile (256x32 bf16) staged by TDM; Q tile cooperative (tiny).
__global__ __launch_bounds__(256)
void scores_kernel(const u16* __restrict__ Qb, const u16* __restrict__ Kpb,
                   const float* __restrict__ P, float* __restrict__ outA){
  __shared__ u32 Qs[32 * 16];
  __shared__ u32 Bs[256 * 16];
  __shared__ float lps[256];
  const int tid = threadIdx.x, lane = tid & 31, wid = tid >> 5;
  const int mw = wid & 1, nw = wid >> 1;            // 2(M) x 4(N)
  const int nBase = blockIdx.x * 256;
  const int b = blockIdx.y;
  const u32* Qu = (const u32*)Qb;
  const u32 bsOff = (u32)(size_t)(void*)Bs;

  { float p = P[b * NN + nBase + tid];
    p = fminf(fmaxf(p, 0.1f), 0.9f);
    lps[tid] = __logf(p); }

  Acc acc[4];
  #pragma unroll
  for (int t = 0; t < 4; ++t)
    #pragma unroll
    for (int j = 0; j < 8; ++j) acc[t].f[j] = 0.f;

  for (int kk = 0; kk < DD; kk += 32){
    if (wid == 0)
      tdm_load_2d_b16(Kpb + (size_t)(b * NN + nBase) * DD + kk, bsOff,
                      DD, DD, DD, 32, DD);
    for (int i = tid; i < 512; i += 256){
      int r = i >> 4, kd = i & 15;
      Qs[i] = Qu[(b * KQ + r) * (DD >> 1) + (kk >> 1) + kd];
    }
    if (wid == 0) __builtin_amdgcn_s_wait_tensorcnt(0);
    __syncthreads();
    Frag a; load_afrag(a, Qs, mw * 16, lane);
    #pragma unroll
    for (int t = 0; t < 4; ++t){
      Frag bfr; load_bfrag(bfr, Bs, nw * 64 + t * 16, lane);
      acc[t].v = wmma_bf16(a, bfr, acc[t].v);
    }
    __syncthreads();
  }

  const float SCALE = 0.0625f; // 256^-0.5
  const int hi = lane >> 4;
  #pragma unroll
  for (int t = 0; t < 4; ++t){
    int nl = nw * 64 + t * 16 + (lane & 15);
    #pragma unroll
    for (int v = 0; v < 8; ++v){
      int m = mw * 16 + v + hi * 8;
      outA[(size_t)b * KQ * NN + (size_t)m * NN + nBase + nl] = acc[t].f[v] * SCALE + lps[nl];
    }
  }
}

__global__ __launch_bounds__(256)
void softmax_kernel(float* __restrict__ A){
  __shared__ float red[256];
  float* row = A + (size_t)blockIdx.x * NN;
  const int tid = threadIdx.x;
  float m = -3.402823466e38f;
  for (int i = tid; i < NN; i += 256) m = fmaxf(m, row[i]);
  red[tid] = m; __syncthreads();
  for (int s = 128; s > 0; s >>= 1){ if (tid < s) red[tid] = fmaxf(red[tid], red[tid + s]); __syncthreads(); }
  float mx = red[0]; __syncthreads();
  float sum = 0.f;
  for (int i = tid; i < NN; i += 256){ float e = __expf(row[i] - mx); row[i] = e; sum += e; }
  red[tid] = sum; __syncthreads();
  for (int s = 128; s > 0; s >>= 1){ if (tid < s) red[tid] += red[tid + s]; __syncthreads(); }
  float inv = 1.f / red[0];
  for (int i = tid; i < NN; i += 256) row[i] *= inv;
}

// F[b, m, d] = sum_n A[b,m,n] * V[b,n,d]; V tile transposed into LDS so K(=n) pairs pack.
__global__ __launch_bounds__(256)
void av_kernel(const float* __restrict__ A, const u16* __restrict__ Vb, u16* __restrict__ Fb){
  __shared__ u32 As[32 * 16];
  __shared__ u32 Vt[128 * 16];   // [d_local][32 n] bf16
  const int tid = threadIdx.x, lane = tid & 31, wid = tid >> 5;
  const int mw = wid & 1, nw = wid >> 1;            // 2(M) x 4(D)
  const int dBase = blockIdx.x * 128;
  const int b = blockIdx.y;
  const u32* Vu = (const u32*)Vb;
  u16* As16 = (u16*)As;
  u16* Vt16 = (u16*)Vt;

  Acc acc[2];
  #pragma unroll
  for (int t = 0; t < 2; ++t)
    #pragma unroll
    for (int j = 0; j < 8; ++j) acc[t].f[j] = 0.f;

  for (int nb = 0; nb < NN; nb += 32){
    for (int i = tid; i < 1024; i += 256){
      int r = i >> 5, n = i & 31;
      As16[i] = f2bf(A[(size_t)b * KQ * NN + (size_t)r * NN + nb + n]);
    }
    for (int i = tid; i < 2048; i += 256){
      int nl = i >> 6, dd = i & 63;
      u32 u = Vu[(size_t)(b * NN + nb + nl) * (DD >> 1) + (dBase >> 1) + dd];
      Vt16[(2 * dd + 0) * 32 + nl] = (u16)(u & 0xFFFFu);
      Vt16[(2 * dd + 1) * 32 + nl] = (u16)(u >> 16);
    }
    __syncthreads();
    Frag a; load_afrag(a, As, mw * 16, lane);
    #pragma unroll
    for (int t = 0; t < 2; ++t){
      Frag bfr; load_bfrag(bfr, Vt, nw * 32 + t * 16, lane);
      acc[t].v = wmma_bf16(a, bfr, acc[t].v);
    }
    __syncthreads();
  }

  const int hi = lane >> 4;
  #pragma unroll
  for (int t = 0; t < 2; ++t){
    int d = dBase + nw * 32 + t * 16 + (lane & 15);
    #pragma unroll
    for (int v = 0; v < 8; ++v){
      int m = mw * 16 + v + hi * 8;
      Fb[(size_t)(b * KQ + m) * DD + d] = f2bf(acc[t].f[v]);
    }
  }
}

__global__ __launch_bounds__(256)
void normalize_kernel(const float* __restrict__ F2, float* __restrict__ out){
  __shared__ float red[256];
  const int row = blockIdx.x, tid = threadIdx.x;
  const float* p = F2 + (size_t)row * DIMM;
  float ss = 0.f;
  for (int i = tid; i < DIMM; i += 256){ float v = p[i]; ss += v * v; }
  red[tid] = ss; __syncthreads();
  for (int s = 128; s > 0; s >>= 1){ if (tid < s) red[tid] += red[tid + s]; __syncthreads(); }
  float inv = 1.f / (sqrtf(red[0]) + 1e-8f);
  for (int i = tid; i < DIMM; i += 256) out[(size_t)row * DIMM + i] = p[i] * inv;
}

extern "C" void kernel_launch(void* const* d_in, const int* in_sizes, int n_in,
                              void* d_out, int out_size, void* d_ws, size_t ws_size,
                              hipStream_t stream){
  (void)in_sizes; (void)n_in; (void)out_size; (void)ws_size;
  const float* E  = (const float*)d_in[0];   // [B,N,DIM]
  const float* T  = (const float*)d_in[1];   // [B,K,DIM]
  const float* P  = (const float*)d_in[2];   // [B,N]
  const float* Wq = (const float*)d_in[3];   // [D,DIM]
  const float* Wk = (const float*)d_in[4];
  const float* Wv = (const float*)d_in[5];
  const float* Wo = (const float*)d_in[6];   // [DIM,D]

  char* ws = (char*)d_ws;
  const size_t WSZ  = (size_t)DD * DIMM * 2;           // 256 KB per bf16 weight
  const size_t KVSZ = (size_t)BB * NN * DD * 2;        // 32 MB per bf16 K/V
  u16* WQb = (u16*)(ws + 0 * WSZ);
  u16* WKb = (u16*)(ws + 1 * WSZ);
  u16* WVb = (u16*)(ws + 2 * WSZ);
  u16* WOb = (u16*)(ws + 3 * WSZ);
  u16* Qb  = (u16*)(ws + 4 * WSZ);                     // [512,256] bf16
  u16* Kpb = (u16*)(ws + 5 * WSZ);                     // [65536,256] bf16
  u16* Vbf = (u16*)(ws + 5 * WSZ + KVSZ);
  u16* Fb  = (u16*)(ws + 5 * WSZ + 2 * KVSZ);          // [512,256] bf16
  float* F2 = (float*)(ws + 5 * WSZ + 2 * KVSZ + WSZ); // [512,512] f32

  float* outF = (float*)d_out;                         // [B,K,DIM]
  float* outA = outF + (size_t)BB * KQ * DIMM;         // [B,K,N]

  cvt_kernel<<<512, 256, 0, stream>>>(Wq, WQb, DD * DIMM);
  cvt_kernel<<<512, 256, 0, stream>>>(Wk, WKb, DD * DIMM);
  cvt_kernel<<<512, 256, 0, stream>>>(Wv, WVb, DD * DIMM);
  cvt_kernel<<<512, 256, 0, stream>>>(Wo, WOb, DIMM * DD);

  // Fused K/V projection: single pass over E, TDM-staged weight tiles
  proj_kv_kernel<<<dim3(1024, 1), 256, 0, stream>>>(E, WKb, WVb, Kpb, Vbf);
  // Q projection
  gemm_xwT<1, 0><<<dim3(8, 1), 256, 0, stream>>>(T, WQb, Qb, DIMM, DD);

  scores_kernel<<<dim3(NN / 256, BB), 256, 0, stream>>>(Qb, Kpb, P, outA);
  softmax_kernel<<<BB * KQ, 256, 0, stream>>>(outA);
  av_kernel<<<dim3(2, BB), 256, 0, stream>>>(outA, Vbf, Fb);

  // out-projection: F @ Wo^T -> fp32, then row L2-normalize into d_out
  gemm_xwT<0, 1><<<dim3(8, 2), 256, 0, stream>>>(Fb, WOb, F2, DD, DIMM);
  normalize_kernel<<<BB * KQ, 256, 0, stream>>>(F2, outF);
}
